// TNPDEncoder_74268574482640
// MI455X (gfx1250) — matrix operand order repaired and checked by reading, not compiled, not run on hardware
//
#include <hip/hip_runtime.h>
#include <math.h>

typedef __attribute__((ext_vector_type(16))) __bf16 v16bf;
typedef __attribute__((ext_vector_type(8)))  float  v8f;

#define MBATCH 8
#define NTOK   1024
#define NCTX   512
#define DMODEL 256
#define NHEAD  8
#define DHEAD  32
#define NLAYER 6
#define DFF    1024
#define TOKENS (MBATCH * NTOK)   // 8192

// ---------- WMMA operand packing helpers ----------
// A-matrix (16x32 bf16): lane holds row m = lane&15; element j -> k = half*16 + j.
// Source: f32 row-major, ld = K.
__device__ __forceinline__ v16bf load_a_f32(const float* A, int lda, int row0, int k0, int lane) {
  int m = lane & 15, half = lane >> 4;
  const float* p0 = A + (size_t)(row0 + m) * lda + k0 + half * 16;
  const float* p1 = p0 + 8;
  v16bf a;
#pragma unroll
  for (int j = 0; j < 8; ++j) { a[j] = (__bf16)p0[j]; a[8 + j] = (__bf16)p1[j]; }
  return a;
}

// B-matrix (32x16 bf16): lane holds column n = lane&15; element j -> k = half*16 + j.
// Source: pre-transposed bf16 weights Wt[N][K] (contiguous in k).
__device__ __forceinline__ v16bf load_b_wt(const __bf16* Wt, int K, int n0, int k0, int lane) {
  int n = n0 + (lane & 15), half = lane >> 4;
  const __bf16* p = Wt + (size_t)n * K + k0 + half * 16;
  v16bf b;
#pragma unroll
  for (int j = 0; j < 16; ++j) b[j] = p[j];
  return b;
}

// ---------- generic GEMM: each wave computes a 16x64 strip ----------
// C[m][n] = act(A[m][:]@Wt[n][:] + bias[n]) (+residual); A loaded/converted once per
// k-step and reused across 4 WMMAs (4x less activation traffic + conversion VALU).
__global__ __launch_bounds__(32)
void gemm_bf16_wmma(const float* __restrict__ A, const __bf16* __restrict__ Wt,
                    const float* __restrict__ bias, const float* __restrict__ residual,
                    float* __restrict__ C, int N, int K, int relu) {
  int lane = threadIdx.x;
  int n0 = blockIdx.x * 64;
  int m0 = blockIdx.y * 16;
  v8f acc[4] = {{}, {}, {}, {}};
  for (int k0 = 0; k0 < K; k0 += 32) {
    v16bf a = load_a_f32(A, K, m0, k0, lane);
#pragma unroll
    for (int i = 0; i < 4; ++i) {
      v16bf b = load_b_wt(Wt, K, n0 + i * 16, k0, lane);
      acc[i] = __builtin_amdgcn_wmma_f32_16x16x32_bf16(false, a, false, b, (short)0, acc[i], false, false);
    }
  }
  int half = lane >> 4;
#pragma unroll
  for (int i = 0; i < 4; ++i) {
    int n = n0 + i * 16 + (lane & 15);
    float bv = bias ? bias[n] : 0.f;
#pragma unroll
    for (int r = 0; r < 8; ++r) {
      int m = m0 + half * 8 + r;
      float v = acc[i][r] + bv;
      if (relu) v = fmaxf(v, 0.f);
      if (residual) v += residual[(size_t)m * N + n];
      C[(size_t)m * N + n] = v;
    }
  }
}

// ---------- LayerNorm: one wave per token ----------
__global__ __launch_bounds__(32)
void layernorm_k(const float* __restrict__ x, const float* __restrict__ g,
                 const float* __restrict__ b, float* __restrict__ y) {
  int t = blockIdx.x, lane = threadIdx.x;
  const float* xr = x + (size_t)t * DMODEL;
  float v[8], s = 0.f;
#pragma unroll
  for (int j = 0; j < 8; ++j) { v[j] = xr[lane + j * 32]; s += v[j]; }
  for (int off = 16; off; off >>= 1) s += __shfl_xor(s, off, 32);
  float mu = s * (1.f / DMODEL);
  float vs = 0.f;
#pragma unroll
  for (int j = 0; j < 8; ++j) { float d = v[j] - mu; vs += d * d; }
  for (int off = 16; off; off >>= 1) vs += __shfl_xor(vs, off, 32);
  float rs = rsqrtf(vs * (1.f / DMODEL) + 1e-5f);
  float* yr = y + (size_t)t * DMODEL;
#pragma unroll
  for (int j = 0; j < 8; ++j) {
    int c = lane + j * 32;
    yr[c] = (v[j] - mu) * rs * g[c] + b[c];
  }
}

// ---------- embed + encoder layer 1 (K=10, plain f32) ----------
__global__ __launch_bounds__(256)
void embed_enc1(const float* __restrict__ xc, const float* __restrict__ yc,
                const float* __restrict__ xt, const float* __restrict__ W1,
                const float* __restrict__ b1, float* __restrict__ h) {
  __shared__ float zl[10];
  int t = blockIdx.x;
  int bidx = t >> 10, i = t & 1023;
  int tid = threadIdx.x;
  if (tid < 10) {
    float v;
    if (i < NCTX) {
      if (tid < 8)       v = xc[((size_t)bidx * NCTX + i) * 8 + tid];
      else if (tid == 8) v = yc[(size_t)bidx * NCTX + i];
      else               v = 0.f;
    } else {
      int it = i - NCTX;
      if (tid < 8)       v = xt[((size_t)bidx * NCTX + it) * 8 + tid];
      else if (tid == 8) v = 0.f;
      else               v = 1.f;
    }
    zl[tid] = v;
  }
  __syncthreads();
  float acc = b1[tid];
#pragma unroll
  for (int k = 0; k < 10; ++k) acc += zl[k] * W1[k * DMODEL + tid];
  h[(size_t)t * DMODEL + tid] = fmaxf(acc, 0.f);
}

// ---------- masked attention: one wave per (batch, head, 16-row tile) ----------
__global__ __launch_bounds__(32)
void attn_wmma(const float* __restrict__ qkv, float* __restrict__ o) {
  __shared__ float s[16][NCTX];   // 32 KB score/prob tile
  __shared__ float invsum[16];
  __shared__ float pself[16];
  int lane = threadIdx.x;
  int t  = blockIdx.x;            // row tile 0..63
  int hh = blockIdx.y;            // head
  int b  = blockIdx.z;            // batch
  int m = lane & 15, half = lane >> 4;
  int row0 = t * 16;              // within batch
  size_t batchBase = (size_t)b * NTOK;
  const int LDQ = 3 * DMODEL;     // 768
  int isTarget = (row0 >= NCTX);
  const float scale = 0.1767766952966369f;  // 1/sqrt(32)

  // Q tile (fixed A operand)
  v16bf aQ;
  {
    const float* qp = qkv + (batchBase + row0 + m) * LDQ + hh * DHEAD;
    const float* p0 = qp + half * 16;
    const float* p1 = p0 + 8;
#pragma unroll
    for (int j = 0; j < 8; ++j) { aQ[j] = (__bf16)p0[j]; aQ[8 + j] = (__bf16)p1[j]; }
  }

  // scores over the 512 context keys: B column n = key, B rows = head dim (contiguous)
  for (int kb = 0; kb < NCTX / 16; ++kb) {
    const float* kp = qkv + (batchBase + kb * 16 + m) * LDQ + DMODEL + hh * DHEAD + half * 16;
    v16bf bK;
#pragma unroll
    for (int j = 0; j < 16; ++j) bK[j] = (__bf16)kp[j];
    v8f sc = {};
    sc = __builtin_amdgcn_wmma_f32_16x16x32_bf16(false, aQ, false, bK, (short)0, sc, false, false);
#pragma unroll
    for (int r = 0; r < 8; ++r) s[half * 8 + r][kb * 16 + m] = sc[r] * scale;
  }
  __syncthreads();

  // exact f32 softmax per row (lanes 0..15 own one row each); targets add a self logit
  if (lane < 16) {
    int row = row0 + lane;
    float sself = 0.f;
    if (isTarget) {
      const float* qp = qkv + (batchBase + row) * LDQ + hh * DHEAD;
      const float* kp = qkv + (batchBase + row) * LDQ + DMODEL + hh * DHEAD;
      float d = 0.f;
#pragma unroll
      for (int j = 0; j < DHEAD; ++j) d += qp[j] * kp[j];
      sself = d * scale;
    }
    float mx = isTarget ? sself : -3.0e38f;
    for (int c = 0; c < NCTX; ++c) mx = fmaxf(mx, s[lane][c]);
    float sum = 0.f;
    for (int c = 0; c < NCTX; ++c) { float e = __expf(s[lane][c] - mx); s[lane][c] = e; sum += e; }
    float ps = isTarget ? __expf(sself - mx) : 0.f;
    pself[lane]  = ps;
    invsum[lane] = 1.f / (sum + ps);
  }
  __syncthreads();

  // P @ V  (dh=32 split into two 16-wide output groups)
  v8f acc0 = {}, acc1 = {};
  for (int kb = 0; kb < NCTX; kb += 32) {
    v16bf aP;
#pragma unroll
    for (int j = 0; j < 8; ++j) {
      aP[j]     = (__bf16)s[m][kb + half * 16 + j];
      aP[8 + j] = (__bf16)s[m][kb + half * 16 + 8 + j];
    }
    const float* vp = qkv + (batchBase + kb + half * 16) * LDQ + 2 * DMODEL + hh * DHEAD;
    v16bf bV0, bV1;
#pragma unroll
    for (int j = 0; j < 16; ++j) {
      bV0[j] = (__bf16)vp[(size_t)j * LDQ + m];
      bV1[j] = (__bf16)vp[(size_t)j * LDQ + 16 + m];
    }
    acc0 = __builtin_amdgcn_wmma_f32_16x16x32_bf16(false, aP, false, bV0, (short)0, acc0, false, false);
    acc1 = __builtin_amdgcn_wmma_f32_16x16x32_bf16(false, aP, false, bV1, (short)0, acc1, false, false);
  }
#pragma unroll
  for (int r = 0; r < 8; ++r) {
    int mr = half * 8 + r;
    int row = row0 + mr;
    const float* vself = qkv + (batchBase + row) * LDQ + 2 * DMODEL + hh * DHEAD;
    float inv = invsum[mr], ps = pself[mr];
    float v0 = acc0[r], v1 = acc1[r];
    if (isTarget) { v0 += ps * vself[m]; v1 += ps * vself[16 + m]; }
    o[(batchBase + row) * DMODEL + hh * DHEAD + m]      = v0 * inv;
    o[(batchBase + row) * DMODEL + hh * DHEAD + 16 + m] = v1 * inv;
  }
}

// ---------- weight transpose f32[K][N] -> bf16[N][K] ----------
__global__ void wprep_t(const float* __restrict__ src, __bf16* __restrict__ dst, int K, int N) {
  int idx = blockIdx.x * blockDim.x + threadIdx.x;
  if (idx < K * N) {
    int k = idx / N, n = idx % N;
    dst[(size_t)n * K + k] = (__bf16)src[idx];
  }
}

__global__ void copy_f32(const float* __restrict__ src, float* __restrict__ dst, int n) {
  int i = blockIdx.x * blockDim.x + threadIdx.x;
  if (i < n) dst[i] = src[i];
}

extern "C" void kernel_launch(void* const* d_in, const int* in_sizes, int n_in,
                              void* d_out, int out_size, void* d_ws, size_t ws_size,
                              hipStream_t stream) {
  const float* xc     = (const float*)d_in[0];
  const float* yc     = (const float*)d_in[1];
  const float* xt     = (const float*)d_in[2];
  const float* enc_W1 = (const float*)d_in[3];
  const float* enc_b1 = (const float*)d_in[4];
  const float* enc_W2 = (const float*)d_in[5];
  const float* enc_b2 = (const float*)d_in[6];
  const float* Wqkv   = (const float*)d_in[7];
  const float* bqkv   = (const float*)d_in[8];
  const float* Wo     = (const float*)d_in[9];
  const float* bo     = (const float*)d_in[10];
  const float* ln1_g  = (const float*)d_in[11];
  const float* ln1_b  = (const float*)d_in[12];
  const float* ln2_g  = (const float*)d_in[13];
  const float* ln2_b  = (const float*)d_in[14];
  const float* Wff1   = (const float*)d_in[15];
  const float* bff1   = (const float*)d_in[16];
  const float* Wff2   = (const float*)d_in[17];
  const float* bff2   = (const float*)d_in[18];

  // workspace layout (~94 MB)
  float* z    = (float*)d_ws;
  float* h    = z    + (size_t)TOKENS * DMODEL;
  float* qkvb = h    + (size_t)TOKENS * DMODEL;
  float* ob   = qkvb + (size_t)TOKENS * 3 * DMODEL;
  float* ffh  = ob   + (size_t)TOKENS * DMODEL;
  __bf16* encW2t = (__bf16*)(ffh + (size_t)TOKENS * DFF);
  __bf16* lw = encW2t + (size_t)DMODEL * DMODEL;
  const size_t perL = (size_t)DMODEL * 3 * DMODEL + (size_t)DMODEL * DMODEL
                    + (size_t)DMODEL * DFF + (size_t)DFF * DMODEL;

  // --- weight prep (bf16 transposed) ---
  wprep_t<<<(DMODEL * DMODEL + 255) / 256, 256, 0, stream>>>(enc_W2, encW2t, DMODEL, DMODEL);
  for (int l = 0; l < NLAYER; ++l) {
    __bf16* wq = lw + l * perL;
    __bf16* wo = wq + (size_t)DMODEL * 3 * DMODEL;
    __bf16* w1 = wo + (size_t)DMODEL * DMODEL;
    __bf16* w2 = w1 + (size_t)DMODEL * DFF;
    wprep_t<<<(DMODEL * 3 * DMODEL + 255) / 256, 256, 0, stream>>>(Wqkv + (size_t)l * DMODEL * 3 * DMODEL, wq, DMODEL, 3 * DMODEL);
    wprep_t<<<(DMODEL * DMODEL + 255) / 256, 256, 0, stream>>>(Wo + (size_t)l * DMODEL * DMODEL, wo, DMODEL, DMODEL);
    wprep_t<<<(DMODEL * DFF + 255) / 256, 256, 0, stream>>>(Wff1 + (size_t)l * DMODEL * DFF, w1, DMODEL, DFF);
    wprep_t<<<(DFF * DMODEL + 255) / 256, 256, 0, stream>>>(Wff2 + (size_t)l * DFF * DMODEL, w2, DFF, DMODEL);
  }

  // --- encoder ---
  embed_enc1<<<TOKENS, DMODEL, 0, stream>>>(xc, yc, xt, enc_W1, enc_b1, h);
  gemm_bf16_wmma<<<dim3(DMODEL / 64, TOKENS / 16), 32, 0, stream>>>(h, encW2t, enc_b2, nullptr, z, DMODEL, DMODEL, 0);

  // --- transformer layers ---
  for (int l = 0; l < NLAYER; ++l) {
    __bf16* wq = lw + l * perL;
    __bf16* wo = wq + (size_t)DMODEL * 3 * DMODEL;
    __bf16* w1 = wo + (size_t)DMODEL * DMODEL;
    __bf16* w2 = w1 + (size_t)DMODEL * DFF;
    layernorm_k<<<TOKENS, 32, 0, stream>>>(z, ln1_g + l * DMODEL, ln1_b + l * DMODEL, h);
    gemm_bf16_wmma<<<dim3(3 * DMODEL / 64, TOKENS / 16), 32, 0, stream>>>(h, wq, bqkv + l * 3 * DMODEL, nullptr, qkvb, 3 * DMODEL, DMODEL, 0);
    attn_wmma<<<dim3(NTOK / 16, NHEAD, MBATCH), 32, 0, stream>>>(qkvb, ob);
    gemm_bf16_wmma<<<dim3(DMODEL / 64, TOKENS / 16), 32, 0, stream>>>(ob, wo, bo + l * DMODEL, z, z, DMODEL, DMODEL, 0);
    layernorm_k<<<TOKENS, 32, 0, stream>>>(z, ln2_g + l * DMODEL, ln2_b + l * DMODEL, h);
    gemm_bf16_wmma<<<dim3(DFF / 64, TOKENS / 16), 32, 0, stream>>>(h, w1, bff1 + l * DFF, nullptr, ffh, DFF, DMODEL, 1);
    gemm_bf16_wmma<<<dim3(DMODEL / 64, TOKENS / 16), 32, 0, stream>>>(ffh, w2, bff2 + l * DMODEL, z, z, DMODEL, DFF, 0);
  }

  copy_f32<<<(TOKENS * DMODEL + 255) / 256, 256, 0, stream>>>(z, (float*)d_out, TOKENS * DMODEL);
}
